// SelectivePSIv2_49950469653079
// MI455X (gfx1250) — matrix operand order repaired and checked by reading, not compiled, not run on hardware
//
#include <hip/hip_runtime.h>
#include <hip/hip_bf16.h>

typedef __attribute__((ext_vector_type(16))) __bf16 v16bf;
typedef __attribute__((ext_vector_type(8)))  float  v8f;

union AFrag { v16bf v; uint4 q[2]; };

#define MODE_F32T      0   // write f32 transposed [b][n][s], +bias
#define MODE_GELU_BF16 1   // write bf16 row-major [m][n], gelu(+bias)
#define MODE_RESID     2   // write f32 row-major, +bias +resid

__device__ __forceinline__ float gelu_exact(float v) {
    return 0.5f * v * (1.0f + erff(v * 0.70710678118654752f));
}
__device__ __forceinline__ float sigmoidf(float v) {
    return 1.0f / (1.0f + expf(-v));
}

// ---------------------------------------------------------------------------
// bf16 WMMA GEMM: C[M,N] = A[M,K] @ Bt[N,K]^T, f32 accumulate.
// Block = 256 threads = 8 waves (2m x 4n); wave tile 32x32 = 2x2 wmma tiles.
// Per K-step: 2 A frags + 2 B frags (4 KB) -> 4 wmma (16 flops/byte).
// ---------------------------------------------------------------------------
__global__ __launch_bounds__(256)
void gemm_wmma_bf16(const __hip_bfloat16* __restrict__ A,
                    const __hip_bfloat16* __restrict__ Bt,
                    const float* __restrict__ bias,
                    float* __restrict__ outF,
                    __hip_bfloat16* __restrict__ outH,
                    const float* __restrict__ resid,
                    int N, int K, int mode)
{
    const int SEQ = 2048;
    int tid  = threadIdx.x;
    int wave = tid >> 5, lane = tid & 31;
    int half = lane >> 4, l16 = lane & 15;
    int tile_m = blockIdx.y * 64  + (wave >> 2) * 32;   // wave owns 32 rows
    int tile_n = blockIdx.x * 128 + (wave & 3) * 32;    // wave owns 32 cols

    const __hip_bfloat16* arow0 = A  + (size_t)(tile_m + l16) * K;
    const __hip_bfloat16* arow1 = arow0 + (size_t)16 * K;
    const __hip_bfloat16* brow0 = Bt + (size_t)(tile_n + l16) * K;
    const __hip_bfloat16* brow1 = brow0 + (size_t)16 * K;
    const int aoff = half * 8;    // A lane-half K offset (ISA 16-bit A 16x32 layout)
    const int boff = half * 8;    // B lane-half K offset (same K order as A)

    v8f acc00 = {}, acc01 = {}, acc10 = {}, acc11 = {};
    for (int k0 = 0; k0 < K; k0 += 32) {
        AFrag a0, a1, b0, b1;
        a0.q[0] = *(const uint4*)(arow0 + k0 + aoff);
        a0.q[1] = *(const uint4*)(arow0 + k0 + 16 + aoff);
        a1.q[0] = *(const uint4*)(arow1 + k0 + aoff);
        a1.q[1] = *(const uint4*)(arow1 + k0 + 16 + aoff);
        b0.q[0] = *(const uint4*)(brow0 + k0 + boff);
        b0.q[1] = *(const uint4*)(brow0 + k0 + boff + 16);
        b1.q[0] = *(const uint4*)(brow1 + k0 + boff);
        b1.q[1] = *(const uint4*)(brow1 + k0 + boff + 16);
        acc00 = __builtin_amdgcn_wmma_f32_16x16x32_bf16(
                    false, a0.v, false, b0.v, (short)0, acc00, false, false);
        acc01 = __builtin_amdgcn_wmma_f32_16x16x32_bf16(
                    false, a0.v, false, b1.v, (short)0, acc01, false, false);
        acc10 = __builtin_amdgcn_wmma_f32_16x16x32_bf16(
                    false, a1.v, false, b0.v, (short)0, acc10, false, false);
        acc11 = __builtin_amdgcn_wmma_f32_16x16x32_bf16(
                    false, a1.v, false, b1.v, (short)0, acc11, false, false);
    }

    v8f accs[2][2] = {{acc00, acc01}, {acc10, acc11}};
    for (int mi = 0; mi < 2; ++mi) {
        for (int ni = 0; ni < 2; ++ni) {
            v8f acc = accs[mi][ni];
            int   ng  = tile_n + ni * 16 + l16;
            float bv  = bias[ng];
            int   mg0 = tile_m + mi * 16 + half * 8;  // C: VGPR r -> m=mg0+r, n=ng

            if (mode == MODE_F32T) {
                int bidx = mg0 >> 11;            // token = b*SEQ + s
                int s    = mg0 & (SEQ - 1);      // 8 consecutive s per lane
                float* p = outF + ((size_t)bidx * N + ng) * SEQ + s;
                *(float4*)(p)     = make_float4(acc[0]+bv, acc[1]+bv, acc[2]+bv, acc[3]+bv);
                *(float4*)(p + 4) = make_float4(acc[4]+bv, acc[5]+bv, acc[6]+bv, acc[7]+bv);
            } else if (mode == MODE_GELU_BF16) {
                for (int r = 0; r < 8; ++r) {
                    float v = gelu_exact(acc[r] + bv);
                    outH[(size_t)(mg0 + r) * N + ng] = __float2bfloat16(v);
                }
            } else { // MODE_RESID
                for (int r = 0; r < 8; ++r) {
                    size_t idx = (size_t)(mg0 + r) * N + ng;
                    outF[idx] = acc[r] + bv + resid[idx];
                }
            }
        }
    }
}

// ---------------------------------------------------------------------------
// Elementwise f32 -> bf16
// ---------------------------------------------------------------------------
__global__ void f32_to_bf16_kernel(const float* __restrict__ in,
                                   __hip_bfloat16* __restrict__ out, size_t n)
{
    size_t i = (size_t)blockIdx.x * blockDim.x + threadIdx.x;
    if (i < n) out[i] = __float2bfloat16(in[i]);
}

// ---------------------------------------------------------------------------
// Weight transpose + bf16 convert: W[K,N] f32 -> Wt[N,K] bf16 (LDS tiled)
// ---------------------------------------------------------------------------
__global__ void w_transpose_bf16(const float* __restrict__ W,
                                 __hip_bfloat16* __restrict__ Wt, int K, int N)
{
    __shared__ float tile[32][33];
    int n0 = blockIdx.x * 32, k0 = blockIdx.y * 32;
    for (int i = threadIdx.y; i < 32; i += 8)
        tile[i][threadIdx.x] = W[(size_t)(k0 + i) * N + n0 + threadIdx.x];
    __syncthreads();
    for (int i = threadIdx.y; i < 32; i += 8)
        Wt[(size_t)(n0 + i) * K + k0 + threadIdx.x] =
            __float2bfloat16(tile[threadIdx.x][i]);
}

// ---------------------------------------------------------------------------
// Batched f32 transpose: in [Z][R][C] -> out [Z][C][R] (LDS tiled)
// ---------------------------------------------------------------------------
__global__ void transpose_f32(const float* __restrict__ in,
                              float* __restrict__ out, int R, int C)
{
    __shared__ float tile[32][33];
    size_t bo = (size_t)blockIdx.z * R * C;
    int r0 = blockIdx.y * 32, c0 = blockIdx.x * 32;
    for (int i = threadIdx.y; i < 32; i += 8)
        tile[i][threadIdx.x] = in[bo + (size_t)(r0 + i) * C + c0 + threadIdx.x];
    __syncthreads();
    for (int i = threadIdx.y; i < 32; i += 8)
        out[bo + (size_t)(c0 + i) * R + r0 + threadIdx.x] = tile[threadIdx.x][i];
}

// ---------------------------------------------------------------------------
// Parallel scan: one block per (b,d) channel; S=2048, 8 elems/thread.
// ---------------------------------------------------------------------------
__device__ __forceinline__ float4 block_scan_inc(float4 v, float4* sh)
{
    int t = threadIdx.x;
    sh[t] = v;
    __syncthreads();
    for (int off = 1; off < 256; off <<= 1) {
        float4 u;
        bool p = (t >= off);
        if (p) u = sh[t - off];
        __syncthreads();
        if (p) {
            float4 w = sh[t];
            w.x += u.x; w.y += u.y; w.z += u.z; w.w += u.w;
            sh[t] = w;
        }
        __syncthreads();
    }
    float4 r = sh[t];
    __syncthreads();
    return r;
}

__global__ __launch_bounds__(256)
void scan_kernel(const float* __restrict__ xT,   const float* __restrict__ pdT,
                 const float* __restrict__ wgT,  const float* __restrict__ magT,
                 const float* __restrict__ qpT,  const float* __restrict__ mixT,
                 const float* __restrict__ int_scale,
                 const float* __restrict__ gate_temp,
                 float* __restrict__ outT)
{
    __shared__ float4 sh[256];
    const int S = 2048;
    int ch = blockIdx.x;                 // b*1024 + d
    int b = ch >> 10, d = ch & 1023;
    size_t base      = (size_t)ch * S;                   // [b][d][s]
    size_t mixr_base = ((size_t)b * 2048 + d) * S;       // mix[:, :D] transposed
    size_t mixi_base = ((size_t)b * 2048 + 1024 + d) * S;
    int t  = threadIdx.x;
    int e0 = t * 8;
    float isc = fabsf(int_scale[d]);
    float gt  = gate_temp[0];

    // ---- Pass 1: phi = inclusive cumsum(phase_delta * |int_scale|) ----
    float phi[8];
    {
        float4 p0 = *(const float4*)(pdT + base + e0);
        float4 p1 = *(const float4*)(pdT + base + e0 + 4);
        float pd[8] = {p0.x,p0.y,p0.z,p0.w, p1.x,p1.y,p1.z,p1.w};
        float run = 0.f;
        for (int j = 0; j < 8; ++j) { run += pd[j] * isc; phi[j] = run; }
        float4 sc = block_scan_inc(make_float4(run, 0.f, 0.f, 0.f), sh);
        float pref = sc.x - run;
        for (int j = 0; j < 8; ++j) phi[j] += pref;
    }

    // ---- Pass 2: terms for memory_real / memory_imag / accum_weight ----
    float cr[8], ci[8], cw[8];
    {
        float4 a0 = *(const float4*)(xT   + base + e0);
        float4 a1 = *(const float4*)(xT   + base + e0 + 4);
        float4 g0 = *(const float4*)(wgT  + base + e0);
        float4 g1 = *(const float4*)(wgT  + base + e0 + 4);
        float4 m0 = *(const float4*)(magT + base + e0);
        float4 m1 = *(const float4*)(magT + base + e0 + 4);
        float xv[8] = {a0.x,a0.y,a0.z,a0.w, a1.x,a1.y,a1.z,a1.w};
        float gv[8] = {g0.x,g0.y,g0.z,g0.w, g1.x,g1.y,g1.z,g1.w};
        float mv[8] = {m0.x,m0.y,m0.z,m0.w, m1.x,m1.y,m1.z,m1.w};
        for (int j = 0; j < 8; ++j) {
            float wg  = sigmoidf(gv[j] * gt);
            float mag = 5.0f * sigmoidf(mv[j]);
            float c   = wg * xv[j] * mag;
            float sp, cp; sincosf(phi[j], &sp, &cp);
            cr[j] = c * cp;
            ci[j] = c * sp;
            cw[j] = wg * mag + 1e-8f;
        }
    }
    float sr = 0.f, si = 0.f, sw = 0.f;
    for (int j = 0; j < 8; ++j) { sr += cr[j]; si += ci[j]; sw += cw[j]; }
    float4 sc = block_scan_inc(make_float4(sr, si, sw, 0.f), sh);
    float pr = sc.x - sr, pi = sc.y - si, pw = sc.z - sw;

    // ---- Pass 3: retrieval + mixing, write out (outT may alias pdT) ----
    {
        float4 q0 = *(const float4*)(qpT + base + e0);
        float4 q1 = *(const float4*)(qpT + base + e0 + 4);
        float4 r0 = *(const float4*)(mixT + mixr_base + e0);
        float4 r1 = *(const float4*)(mixT + mixr_base + e0 + 4);
        float4 i0 = *(const float4*)(mixT + mixi_base + e0);
        float4 i1 = *(const float4*)(mixT + mixi_base + e0 + 4);
        float qv [8] = {q0.x,q0.y,q0.z,q0.w, q1.x,q1.y,q1.z,q1.w};
        float mrv[8] = {r0.x,r0.y,r0.z,r0.w, r1.x,r1.y,r1.z,r1.w};
        float miv[8] = {i0.x,i0.y,i0.z,i0.w, i1.x,i1.y,i1.z,i1.w};
        float accr = pr, acci = pi, accw = pw;
        float o[8];
        for (int j = 0; j < 8; ++j) {
            accr += cr[j]; acci += ci[j]; accw += cw[j];
            float inv = rsqrtf(accw);
            float mr = accr * inv, mi = acci * inv;
            float q  = phi[j] + qv[j];
            float sq, cq; sincosf(q, &sq, &cq);
            float rre = mr * cq + mi * sq;
            float rim = mi * cq - mr * sq;
            o[j] = sigmoidf(mrv[j]) * rre + sigmoidf(miv[j]) * rim;
        }
        *(float4*)(outT + base + e0)     = make_float4(o[0], o[1], o[2], o[3]);
        *(float4*)(outT + base + e0 + 4) = make_float4(o[4], o[5], o[6], o[7]);
    }
}

// ---------------------------------------------------------------------------
// LayerNorm over D=1024 per token row, bf16 output (feeds next GEMM).
// ---------------------------------------------------------------------------
__global__ __launch_bounds__(256)
void layernorm_bf16_kernel(const float* __restrict__ in,
                           const float* __restrict__ g,
                           const float* __restrict__ bb,
                           __hip_bfloat16* __restrict__ out)
{
    const int D = 1024;
    __shared__ float s1[256], s2[256];
    int row = blockIdx.x, t = threadIdx.x;
    float4 v = ((const float4*)(in + (size_t)row * D))[t];
    s1[t] = v.x + v.y + v.z + v.w;
    s2[t] = v.x*v.x + v.y*v.y + v.z*v.z + v.w*v.w;
    __syncthreads();
    for (int off = 128; off > 0; off >>= 1) {
        if (t < off) { s1[t] += s1[t + off]; s2[t] += s2[t + off]; }
        __syncthreads();
    }
    float mean = s1[0] * (1.0f / D);
    float var  = s2[0] * (1.0f / D) - mean * mean;
    float inv  = rsqrtf(var + 1e-5f);
    int d0 = t * 4;
    float vv[4] = {v.x, v.y, v.z, v.w};
    for (int j = 0; j < 4; ++j)
        out[(size_t)row * D + d0 + j] =
            __float2bfloat16((vv[j] - mean) * inv * g[d0 + j] + bb[d0 + j]);
}

// ---------------------------------------------------------------------------
extern "C" void kernel_launch(void* const* d_in, const int* in_sizes, int n_in,
                              void* d_out, int out_size, void* d_ws, size_t ws_size,
                              hipStream_t stream)
{
    (void)in_sizes; (void)n_in; (void)out_size; (void)ws_size;
    const int D = 1024, S = 2048, Bn = 2, M = Bn * S;  // M = 4096

    const float* x       = (const float*)d_in[0];
    const float* w_pt1   = (const float*)d_in[1];
    const float* b_pt1   = (const float*)d_in[2];
    const float* w_pt2   = (const float*)d_in[3];
    const float* b_pt2   = (const float*)d_in[4];
    const float* i_scale = (const float*)d_in[5];
    const float* w_wg    = (const float*)d_in[6];
    const float* b_wg    = (const float*)d_in[7];
    const float* g_temp  = (const float*)d_in[8];
    const float* w_mag   = (const float*)d_in[9];
    const float* b_mag   = (const float*)d_in[10];
    const float* w_q1    = (const float*)d_in[11];
    const float* b_q1    = (const float*)d_in[12];
    const float* w_q2    = (const float*)d_in[13];
    const float* b_q2    = (const float*)d_in[14];
    const float* w_mix   = (const float*)d_in[15];
    const float* b_mix   = (const float*)d_in[16];
    const float* ln_g    = (const float*)d_in[17];
    const float* ln_b    = (const float*)d_in[18];
    const float* w_o1    = (const float*)d_in[19];
    const float* b_o1    = (const float*)d_in[20];
    const float* w_o2    = (const float*)d_in[21];
    const float* b_o2    = (const float*)d_in[22];
    float* out = (float*)d_out;

    char* ws = (char*)d_ws;
    size_t off = 0;
    auto alloc = [&](size_t bytes) -> char* {
        char* p = ws + off;
        off += (bytes + 255) & ~(size_t)255;
        return p;
    };
    typedef __hip_bfloat16 bf16;
    // bf16 transposed weights (24 MB)
    bf16* wpt1t = (bf16*)alloc((size_t)D * 2*D * 2);
    bf16* wpt2t = (bf16*)alloc((size_t)2*D * D * 2);
    bf16* wwgt  = (bf16*)alloc((size_t)D * D * 2);
    bf16* wmagt = (bf16*)alloc((size_t)D * D * 2);
    bf16* wq1t  = (bf16*)alloc((size_t)D * D * 2);
    bf16* wq2t  = (bf16*)alloc((size_t)D * D * 2);
    bf16* wmixt = (bf16*)alloc((size_t)D * 2*D * 2);
    bf16* wo1t  = (bf16*)alloc((size_t)D * D * 2);
    bf16* wo2t  = (bf16*)alloc((size_t)D * D * 2);
    // activations
    bf16*  xbf  = (bf16*)alloc((size_t)M * D * 2);      // later reused as ln_bf
    bf16*  h1   = (bf16*)alloc((size_t)M * 2*D * 2);    // later q1h / o1h
    float* xT   = (float*)alloc((size_t)M * D * 4);     // later reused as retrieved
    float* pdT  = (float*)alloc((size_t)M * D * 4);     // scan output aliases this
    float* wgT  = (float*)alloc((size_t)M * D * 4);
    float* magT = (float*)alloc((size_t)M * D * 4);
    float* qpT  = (float*)alloc((size_t)M * D * 4);
    float* mixT = (float*)alloc((size_t)M * 2*D * 4);
    bf16*  q1h  = h1;                                    // 8 MB into 16 MB region
    bf16*  o1h  = h1;
    float* outT = pdT;                                   // alias: safe (see scan)
    float* retr = xT;                                    // alias: xT dead post-scan
    bf16*  lnbf = xbf;                                   // alias: xbf dead post-GEMM7

    dim3 tb(32, 8);
    // weight transposes (f32 -> bf16, [K,N] -> [N,K])
    w_transpose_bf16<<<dim3(2*D/32, D/32),   tb, 0, stream>>>(w_pt1, wpt1t, D,   2*D);
    w_transpose_bf16<<<dim3(D/32,   2*D/32), tb, 0, stream>>>(w_pt2, wpt2t, 2*D, D);
    w_transpose_bf16<<<dim3(D/32,   D/32),   tb, 0, stream>>>(w_wg,  wwgt,  D,   D);
    w_transpose_bf16<<<dim3(D/32,   D/32),   tb, 0, stream>>>(w_mag, wmagt, D,   D);
    w_transpose_bf16<<<dim3(D/32,   D/32),   tb, 0, stream>>>(w_q1,  wq1t,  D,   D);
    w_transpose_bf16<<<dim3(D/32,   D/32),   tb, 0, stream>>>(w_q2,  wq2t,  D,   D);
    w_transpose_bf16<<<dim3(2*D/32, D/32),   tb, 0, stream>>>(w_mix, wmixt, D,   2*D);
    w_transpose_bf16<<<dim3(D/32,   D/32),   tb, 0, stream>>>(w_o1,  wo1t,  D,   D);
    w_transpose_bf16<<<dim3(D/32,   D/32),   tb, 0, stream>>>(w_o2,  wo2t,  D,   D);

    // x -> bf16 ; x -> transposed f32 for the scan
    size_t nx = (size_t)M * D;
    f32_to_bf16_kernel<<<dim3((unsigned)((nx + 255) / 256)), 256, 0, stream>>>(x, xbf, nx);
    transpose_f32<<<dim3(D/32, S/32, Bn), tb, 0, stream>>>(x, xT, S, D);

    // --- WMMA GEMMs (block tile 64x128, 8 waves, wave tile 32x32) ---
    // 1: h1 = gelu(x @ w_pt1 + b_pt1)                       [M, 2D] bf16
    gemm_wmma_bf16<<<dim3(2*D/128, M/64), 256, 0, stream>>>(
        xbf, wpt1t, b_pt1, nullptr, h1, nullptr, 2*D, D, MODE_GELU_BF16);
    // 2: phase_delta^T = h1 @ w_pt2 + b_pt2                 [B,D,S] f32
    gemm_wmma_bf16<<<dim3(D/128, M/64), 256, 0, stream>>>(
        h1, wpt2t, b_pt2, pdT, nullptr, nullptr, D, 2*D, MODE_F32T);
    // 3: wg_lin^T = x @ w_wg + b_wg
    gemm_wmma_bf16<<<dim3(D/128, M/64), 256, 0, stream>>>(
        xbf, wwgt, b_wg, wgT, nullptr, nullptr, D, D, MODE_F32T);
    // 4: mag_lin^T = x @ w_mag + b_mag
    gemm_wmma_bf16<<<dim3(D/128, M/64), 256, 0, stream>>>(
        xbf, wmagt, b_mag, magT, nullptr, nullptr, D, D, MODE_F32T);
    // 5: q1h = gelu(x @ w_q1 + b_q1)                        [M, D] bf16
    gemm_wmma_bf16<<<dim3(D/128, M/64), 256, 0, stream>>>(
        xbf, wq1t, b_q1, nullptr, q1h, nullptr, D, D, MODE_GELU_BF16);
    // 6: qp_add^T = q1h @ w_q2 + b_q2
    gemm_wmma_bf16<<<dim3(D/128, M/64), 256, 0, stream>>>(
        q1h, wq2t, b_q2, qpT, nullptr, nullptr, D, D, MODE_F32T);
    // 7: mix^T = x @ w_mix + b_mix                          [B,2D,S] f32
    gemm_wmma_bf16<<<dim3(2*D/128, M/64), 256, 0, stream>>>(
        xbf, wmixt, b_mix, mixT, nullptr, nullptr, 2*D, D, MODE_F32T);

    // --- per-channel parallel scans + retrieval + mixing ---
    scan_kernel<<<dim3(Bn * D), 256, 0, stream>>>(
        xT, pdT, wgT, magT, qpT, mixT, i_scale, g_temp, outT);

    // retrieved^T [B,D,S] -> retrieved [B,S,D]
    transpose_f32<<<dim3(S/32, D/32, Bn), tb, 0, stream>>>(outT, retr, D, S);

    // LayerNorm -> bf16
    layernorm_bf16_kernel<<<dim3(M), 256, 0, stream>>>(retr, ln_g, ln_b, lnbf);

    // 8: o1h = gelu(ln @ w_o1 + b_o1)
    gemm_wmma_bf16<<<dim3(D/128, M/64), 256, 0, stream>>>(
        lnbf, wo1t, b_o1, nullptr, o1h, nullptr, D, D, MODE_GELU_BF16);
    // 9: out = x + (o1h @ w_o2 + b_o2)
    gemm_wmma_bf16<<<dim3(D/128, M/64), 256, 0, stream>>>(
        o1h, wo2t, b_o2, out, nullptr, x, D, D, MODE_RESID);
}
